// GlobalCrossAttention_48163763258070
// MI455X (gfx1250) — matrix operand order repaired and checked by reading, not compiled, not run on hardware
//
#include <hip/hip_runtime.h>

#define BB    2
#define NQ    900
#define NQP   912      // 57 tiles of 16
#define EDIM  256
#define NH    8
#define HD    32
#define LK    4096     // H*W = 64*64
#define QTILES 57

typedef __attribute__((ext_vector_type(16))) _Float16 v16h;
typedef __attribute__((ext_vector_type(4)))  _Float16 v4h;
typedef __attribute__((ext_vector_type(8)))  float    v8f;

__device__ __forceinline__ v8f wmma_f16(v16h a, v16h b, v8f c) {
  // D = A(16x32 f16) * B(32x16 f16) + C(16x16 f32)
  return __builtin_amdgcn_wmma_f32_16x16x32_f16(false, a, false, b, (short)0, c,
                                                false, false);
}

// ---------------------------------------------------------------------------
// f32 -> f16 row-major conversion (vectorized x4)
// ---------------------------------------------------------------------------
__global__ __launch_bounds__(256)
void cvt_kernel(const float* __restrict__ src, _Float16* __restrict__ dst, int n4)
{
  const int i = blockIdx.x * blockDim.x + threadIdx.x;
  if (i >= n4) return;
  const float4 v = ((const float4*)src)[i];
  v4h h;
  h[0] = (_Float16)v.x; h[1] = (_Float16)v.y;
  h[2] = (_Float16)v.z; h[3] = (_Float16)v.w;
  ((v4h*)dst)[i] = h;
}

// ---------------------------------------------------------------------------
// Weight prep: W[k][n] f32 (256x256) -> Wt[n][k] f16, so that WMMA B-fragments
// become contiguous 32B per-lane loads.
// ---------------------------------------------------------------------------
__global__ __launch_bounds__(256)
void wtrans_kernel(const float* __restrict__ W, _Float16* __restrict__ Wt)
{
  const int idx = blockIdx.x * 256 + threadIdx.x;   // 65536 elements
  const int k = idx >> 8, n = idx & 255;
  Wt[(size_t)n * EDIM + k] = (_Float16)W[idx];
}

// ---------------------------------------------------------------------------
// 256-K projection GEMM tile kernel (f16 inputs, f16 transposed weights).
//   mode 0: f16 output, head layout  [b][h][rowPad][d]
//   mode 1: f16 output, V-transposed [b][h][d][LK]
//   mode 2: f32 output, row-major    [b][row][256]  (final projection)
// Block = 128 threads (4 waves); wave owns a 16x16 tile, 8 WMMA K-steps.
// All A/B fragment loads are contiguous 32B per lane.
// ---------------------------------------------------------------------------
__global__ __launch_bounds__(128)
void proj_kernel(const _Float16* __restrict__ X, int inPad,
                 const _Float16* __restrict__ Wt, const float* __restrict__ bias,
                 void* __restrict__ dstv, int M, int outPad, float scale, int mode)
{
  const int lane = threadIdx.x & 31;
  const int wave = threadIdx.x >> 5;
  const int b    = blockIdx.z;
  const int mt   = blockIdx.x;
  const bool hi  = lane >= 16;
  const int  ml  = lane & 15;

  int row = mt * 16 + ml;
  if (row >= M) row = M - 1;                 // clamp (keeps EXEC all-ones)
  const int col = blockIdx.y * 64 + wave * 16 + ml;

  const int ka = hi ? 8 : 0;                 // unused; A-fragment also uses kb
  const int kb = hi ? 16 : 0;                // B-fragment K half select

  const _Float16* xr0 = X  + ((size_t)b * inPad + row) * EDIM;
  const _Float16* wr0 = Wt + (size_t)col * EDIM;

  v8f acc = {};
  #pragma unroll
  for (int kt = 0; kt < 8; ++kt) {
    const int kbase = kt * 32;
    v16h a, bm;
    const _Float16* xr = xr0 + kbase;
    #pragma unroll
    for (int i = 0; i < 16; ++i) a[i] = xr[kb + i];
    const _Float16* wr = wr0 + kbase + kb;
    #pragma unroll
    for (int i = 0; i < 16; ++i) bm[i] = wr[i];
    acc = wmma_f16(a, bm, acc);
  }

  const float bv = bias[col];
  const int h = col >> 5, d = col & 31;
  #pragma unroll
  for (int r = 0; r < 8; ++r) {
    const int orow = mt * 16 + r + (hi ? 8 : 0);
    const float v = (acc[r] + bv) * scale;
    if (mode == 0) {
      if (orow < outPad)
        ((_Float16*)dstv)[((size_t)(b * NH + h) * outPad + orow) * HD + d] = (_Float16)v;
    } else if (mode == 1) {
      ((_Float16*)dstv)[((size_t)(b * NH + h) * HD + d) * LK + orow] = (_Float16)v;
    } else {
      if (orow < M)
        ((float*)dstv)[((size_t)b * M + orow) * EDIM + col] = v;
    }
  }
}

// ---------------------------------------------------------------------------
// Box-RPE MLP: per (b,q,pos) compute 2 -> 512(relu) -> 8 heads.
// out layout: [b][q][pos][8] fp32.  K=2 first layer => VALU, not WMMA.
// ---------------------------------------------------------------------------
__device__ __forceinline__ float log_delta(float d) {
  return copysignf(__log2f(fabsf(d) + 1.0f) * (1.0f / 3.0f), d);
}

__global__ __launch_bounds__(128)
void rpe_kernel(const float* __restrict__ rp, const float* __restrict__ W1,
                const float* __restrict__ b1, const float* __restrict__ W2,
                float* __restrict__ out, int ic, int iw)
{
  const int idx = blockIdx.x * blockDim.x + threadIdx.x;
  if (idx >= BB * NQ * 64) return;
  const int p = idx & 63;
  const int t = idx >> 6;                       // b*NQ + q
  const float c  = rp[(size_t)t * 4 + ic];
  const float w  = rp[(size_t)t * 4 + iw];
  const float e1 = (c - 0.5f * w) * 1024.0f;    // (W*STRIDE)=1024
  const float e2 = (c + 0.5f * w) * 1024.0f;
  const float pos = ((float)p + 0.5f) * 16.0f;
  const float d1 = log_delta(e1 - pos);
  const float d2 = log_delta(e2 - pos);

  float acc[NH];
  #pragma unroll
  for (int hh = 0; hh < NH; ++hh) acc[hh] = 0.0f;

  for (int j = 0; j < 512; ++j) {
    float hv = fmaf(d1, W1[j], fmaf(d2, W1[512 + j], b1[j]));
    hv = fmaxf(hv, 0.0f);
    const float* w2 = W2 + (size_t)j * NH;
    #pragma unroll
    for (int hh = 0; hh < NH; ++hh) acc[hh] = fmaf(hv, w2[hh], acc[hh]);
  }
  float* o = out + (size_t)idx * NH;
  #pragma unroll
  for (int hh = 0; hh < NH; ++hh) o[hh] = acc[hh];
}

// ---------------------------------------------------------------------------
// Fused flash attention: one wave per (b, head, 16-query tile).
// Key sweep is y-major (64 keys per image row):
//   - rpe_x values (only 4 distinct per lane!) preloaded once into registers
//   - rpe_y values loaded once per y-row (8 scalars)
//   - per 64 keys: 4 S-WMMAs -> one online-softmax pass -> P via LDS -> 4 PV-WMMAs
// Never materializes the 4096-wide logits.
// ---------------------------------------------------------------------------
__global__ __launch_bounds__(32)
void attn_kernel(const _Float16* __restrict__ Qh, const _Float16* __restrict__ Kh,
                 const _Float16* __restrict__ Vt, const float* __restrict__ rpeX,
                 const float* __restrict__ rpeY, _Float16* __restrict__ Oh)
{
  __shared__ _Float16 ldsP[16 * 64];          // one 16x64 P tile per wave/block

  const int lane = threadIdx.x & 31;
  const bool hi  = lane >= 16;
  const int  ml  = lane & 15;
  const int  ka  = hi ? 8 : 0;
  const int  kb16 = hi ? 16 : 0;
  const int  colk = ml;

  int bid = blockIdx.x;
  const int qt = bid % QTILES; bid /= QTILES;
  const int h  = bid % NH;
  const int b  = bid / NH;

  // Q A-fragment (16 queries x 32 dims), resident for whole sweep
  const _Float16* qbase = Qh + ((size_t)(b * NH + h) * NQP + qt * 16) * HD;
  v16h aq;
  {
    const _Float16* qr = qbase + (size_t)ml * HD;
    #pragma unroll
    for (int i = 0; i < 16; ++i) aq[i] = qr[kb16 + i];
  }

  const _Float16* kbaseh = Kh + (size_t)(b * NH + h) * LK * HD;   // [key][d]
  const _Float16* vbase  = Vt + (size_t)(b * NH + h) * HD * LK;   // [d][key]
  const float* ry = rpeY + (size_t)b * NQ * 64 * NH + h;
  const float* rx = rpeX + (size_t)b * NQ * 64 * NH + h;

  size_t qoff[8];
  #pragma unroll
  for (int r = 0; r < 8; ++r) {
    int q = qt * 16 + r + (hi ? 8 : 0);
    if (q >= NQ) q = NQ - 1;                   // pad rows: clamp rpe gather
    qoff[r] = (size_t)q * 64 * NH;
  }

  // preload rpe_x: per lane only x = xt*16 + colk ever occurs (xt = 0..3)
  float pxv[8][4];
  #pragma unroll
  for (int r = 0; r < 8; ++r)
    #pragma unroll
    for (int xt = 0; xt < 4; ++xt)
      pxv[r][xt] = rx[qoff[r] + (size_t)(xt * 16 + colk) * NH];

  float mrow[8], lrow[8];
  #pragma unroll
  for (int r = 0; r < 8; ++r) { mrow[r] = -INFINITY; lrow[r] = 0.0f; }
  v8f o0 = {}, o1 = {};

  for (int y = 0; y < 64; ++y) {
    const int kyb = y * 64;

    // prefetch next key-row of K and V (gfx1250 global_prefetch path)
    if (y < 63) {
      __builtin_prefetch(kbaseh + (size_t)(kyb + 64 + lane) * HD, 0, 3);
      __builtin_prefetch(vbase + (size_t)lane * LK + kyb + 64, 0, 3);
    }

    // rpe_y for this image row (uniform across the 16-key tiles)
    float ryv[8];
    #pragma unroll
    for (int r = 0; r < 8; ++r) ryv[r] = ry[qoff[r] + (size_t)y * NH];

    // ---- S = Q @ K^T for 64 keys (4 WMMA subtiles) ----
    v8f s[4];
    #pragma unroll
    for (int xt = 0; xt < 4; ++xt) {
      v16h bk;
      const _Float16* kr = kbaseh + (size_t)(kyb + xt * 16 + colk) * HD + kb16;
      #pragma unroll
      for (int i = 0; i < 16; ++i) bk[i] = kr[i];
      v8f z = {};
      s[xt] = wmma_f16(aq, bk, z);
    }

    // ---- add separable RPE bias (all from registers) ----
    #pragma unroll
    for (int r = 0; r < 8; ++r)
      #pragma unroll
      for (int xt = 0; xt < 4; ++xt)
        s[xt][r] += ryv[r] + pxv[r][xt];

    // ---- online softmax update + stage P into LDS ----
    #pragma unroll
    for (int r = 0; r < 8; ++r) {
      float v = fmaxf(fmaxf(s[0][r], s[1][r]), fmaxf(s[2][r], s[3][r]));
      v = fmaxf(v, __shfl_xor(v, 1, 32));
      v = fmaxf(v, __shfl_xor(v, 2, 32));
      v = fmaxf(v, __shfl_xor(v, 4, 32));
      v = fmaxf(v, __shfl_xor(v, 8, 32));
      const float mn = fmaxf(mrow[r], v);
      const float al = __expf(mrow[r] - mn);
      mrow[r] = mn;
      const int prow = r + (hi ? 8 : 0);
      float ps = 0.0f;
      #pragma unroll
      for (int xt = 0; xt < 4; ++xt) {
        const float p = __expf(s[xt][r] - mn);
        ldsP[prow * 64 + xt * 16 + colk] = (_Float16)p;
        ps += p;
      }
      ps += __shfl_xor(ps, 1, 32);
      ps += __shfl_xor(ps, 2, 32);
      ps += __shfl_xor(ps, 4, 32);
      ps += __shfl_xor(ps, 8, 32);
      lrow[r] = lrow[r] * al + ps;
      o0[r] *= al;
      o1[r] *= al;
    }

    // ---- O += P @ V : K=64 keys as two K=32 WMMA chunks x two d-halves ----
    #pragma unroll
    for (int kc = 0; kc < 2; ++kc) {
      v16h ap;
      const _Float16* pr = &ldsP[ml * 64 + kc * 32];
      #pragma unroll
      for (int i = 0; i < 16; ++i) ap[i] = pr[kb16 + i];

      v16h bv;
      const _Float16* vr = vbase + (size_t)colk * LK + kyb + kc * 32 + kb16;
      #pragma unroll
      for (int i = 0; i < 16; ++i) bv[i] = vr[i];
      o0 = wmma_f16(ap, bv, o0);

      vr = vbase + (size_t)(16 + colk) * LK + kyb + kc * 32 + kb16;
      #pragma unroll
      for (int i = 0; i < 16; ++i) bv[i] = vr[i];
      o1 = wmma_f16(ap, bv, o1);
    }
  }

  // ---- normalize + store f16 attention output [b][NQP][256] ----
  _Float16* ob = Oh + ((size_t)b * NQP + qt * 16) * EDIM + h * HD;
  #pragma unroll
  for (int r = 0; r < 8; ++r) {
    const int prow = r + (hi ? 8 : 0);
    const float inv = 1.0f / lrow[r];
    ob[(size_t)prow * EDIM + colk]      = (_Float16)(o0[r] * inv);
    ob[(size_t)prow * EDIM + 16 + colk] = (_Float16)(o1[r] * inv);
  }
}

// ---------------------------------------------------------------------------
extern "C" void kernel_launch(void* const* d_in, const int* in_sizes, int n_in,
                              void* d_out, int out_size, void* d_ws, size_t ws_size,
                              hipStream_t stream)
{
  (void)in_sizes; (void)n_in; (void)out_size; (void)ws_size;

  const float* query = (const float*)d_in[0];
  const float* key   = (const float*)d_in[1];
  const float* value = (const float*)d_in[2];
  const float* rp    = (const float*)d_in[3];
  const float* Wq = (const float*)d_in[4];   const float* bq = (const float*)d_in[5];
  const float* Wk = (const float*)d_in[6];   const float* bk = (const float*)d_in[7];
  const float* Wv = (const float*)d_in[8];   const float* bv = (const float*)d_in[9];
  const float* Wo = (const float*)d_in[10];  const float* bo = (const float*)d_in[11];
  const float* W1x = (const float*)d_in[12]; const float* b1x = (const float*)d_in[13];
  const float* W2x = (const float*)d_in[14];
  const float* W1y = (const float*)d_in[15]; const float* b1y = (const float*)d_in[16];
  const float* W2y = (const float*)d_in[17];

  // workspace carving (256B aligned)
  char* ws = (char*)d_ws;
  size_t off = 0;
  auto carve = [&](size_t bytes) -> char* {
    char* p = ws + off;
    off += (bytes + 255) & ~(size_t)255;
    return p;
  };
  _Float16* Qh    = (_Float16*)carve(sizeof(_Float16) * (size_t)BB * NH * NQP * HD);
  _Float16* Kh    = (_Float16*)carve(sizeof(_Float16) * (size_t)BB * NH * LK  * HD);
  _Float16* Vt    = (_Float16*)carve(sizeof(_Float16) * (size_t)BB * NH * HD  * LK);
  float*    rpeX  = (float*)   carve(sizeof(float)    * (size_t)BB * NQ * 64 * NH);
  float*    rpeY  = (float*)   carve(sizeof(float)    * (size_t)BB * NQ * 64 * NH);
  _Float16* attnO = (_Float16*)carve(sizeof(_Float16) * (size_t)BB * NQP * EDIM);
  _Float16* Qx    = (_Float16*)carve(sizeof(_Float16) * (size_t)BB * NQ * EDIM);
  _Float16* Kx    = (_Float16*)carve(sizeof(_Float16) * (size_t)BB * LK * EDIM);
  _Float16* Vx    = (_Float16*)carve(sizeof(_Float16) * (size_t)BB * LK * EDIM);
  _Float16* WqT   = (_Float16*)carve(sizeof(_Float16) * (size_t)EDIM * EDIM);
  _Float16* WkT   = (_Float16*)carve(sizeof(_Float16) * (size_t)EDIM * EDIM);
  _Float16* WvT   = (_Float16*)carve(sizeof(_Float16) * (size_t)EDIM * EDIM);
  _Float16* WoT   = (_Float16*)carve(sizeof(_Float16) * (size_t)EDIM * EDIM);

  const float scale = 0.17677669529663687f;   // head_dim^-0.5, fused into Q

  // one-time f32->f16 conversions / weight transposes
  const int nq4 = BB * NQ * EDIM / 4, nk4 = BB * LK * EDIM / 4;
  cvt_kernel<<<(nq4 + 255) / 256, 256, 0, stream>>>(query, Qx, nq4);
  cvt_kernel<<<(nk4 + 255) / 256, 256, 0, stream>>>(key,   Kx, nk4);
  cvt_kernel<<<(nk4 + 255) / 256, 256, 0, stream>>>(value, Vx, nk4);
  wtrans_kernel<<<256, 256, 0, stream>>>(Wq, WqT);
  wtrans_kernel<<<256, 256, 0, stream>>>(Wk, WkT);
  wtrans_kernel<<<256, 256, 0, stream>>>(Wv, WvT);
  wtrans_kernel<<<256, 256, 0, stream>>>(Wo, WoT);

  // Q/K/V projections (WMMA f16, fp32 accumulate)
  proj_kernel<<<dim3(QTILES, 4, BB), 128, 0, stream>>>(Qx, NQ, WqT, bq, Qh, NQ,  NQP, scale, 0);
  proj_kernel<<<dim3(LK/16,  4, BB), 128, 0, stream>>>(Kx, LK, WkT, bk, Kh, LK,  LK,  1.0f,  0);
  proj_kernel<<<dim3(LK/16,  4, BB), 128, 0, stream>>>(Vx, LK, WvT, bv, Vt, LK,  LK,  1.0f,  1);

  // separable box-RPE tables
  const int nrpe = BB * NQ * 64;
  rpe_kernel<<<(nrpe + 127) / 128, 128, 0, stream>>>(rp, W1x, b1x, W2x, rpeX, 0, 2);
  rpe_kernel<<<(nrpe + 127) / 128, 128, 0, stream>>>(rp, W1y, b1y, W2y, rpeY, 1, 3);

  // fused flash attention (QK^T + rpe + online softmax + PV)
  attn_kernel<<<BB * NH * QTILES, 32, 0, stream>>>(Qh, Kh, Vt, rpeX, rpeY, attnO);

  // output projection -> fp32 d_out [2,900,256]
  proj_kernel<<<dim3(QTILES, 4, BB), 128, 0, stream>>>(attnO, NQP, WoT, bo, d_out, NQ, NQ, 1.0f, 2);
}
